// SAGEConvReconstructionModule_36850819400182
// MI455X (gfx1250) — matrix operand order repaired and checked by reading, not compiled, not run on hardware
//
#include <hip/hip_runtime.h>
#include <hip/hip_bf16.h>

// SAGEConv x2 for MI455X (gfx1250, wave32).
//  - Aggregation commutes with linear projection: project first (WMMA bf16->f32),
//    then scatter-add 64/32-wide rows (L2-resident, fp32 atomics).
//  - GEMMs use v_wmma_f32_16x16x32_bf16; fragment layouts per CDNA5 ISA 7.12.2.
//  - A tiles staged into LDS via the Tensor Data Mover (tensor_load_to_lds +
//    s_wait_tensorcnt) when the builtin exists; cooperative copy otherwise.

typedef __attribute__((ext_vector_type(16))) __bf16        v16bf;
typedef __attribute__((ext_vector_type(8)))  float         v8f;
typedef __attribute__((ext_vector_type(4)))  unsigned int  u32x4;
typedef __attribute__((ext_vector_type(8)))  int           i32x8;
typedef __attribute__((ext_vector_type(4)))  int           i32x4;

union Frag16 {
    unsigned int u[8];
    uint4        q[2];
    v16bf        v;
};

__device__ __forceinline__ unsigned short f2bf(float f) {
    unsigned int u = __float_as_uint(f);
    unsigned int r = u + 0x7FFFu + ((u >> 16) & 1u);   // round-to-nearest-even
    return (unsigned short)(r >> 16);
}

__device__ __forceinline__ float lrelu(float v) {
    return v > 0.0f ? v : 0.1f * v;
}

// ---------------- degree ----------------
__global__ void deg_kernel(const int* __restrict__ ei, float* __restrict__ deg, int E) {
    int e = blockIdx.x * blockDim.x + threadIdx.x;
    if (e < E) atomicAdd(&deg[ei[E + e]], 1.0f);
}

__global__ void inv_kernel(float* __restrict__ deg, int n) {
    int i = blockIdx.x * blockDim.x + threadIdx.x;
    if (i < n) deg[i] = 1.0f / fmaxf(deg[i], 1.0f);
}

// ---------------- bf16 conversions ----------------
__global__ void cvt_x_kernel(const float* __restrict__ x, unsigned short* __restrict__ xb, int total) {
    int i = blockIdx.x * blockDim.x + threadIdx.x;
    if (i < total) xb[i] = f2bf(x[i]);
}

// WT1: [128 cols][128 k] col-major bf16 of [W_l1 | W_r1]; WT2: [64 cols][64 k] of [W_l2 | W_r2]
__global__ void cvt_w_kernel(const float* __restrict__ Wl1, const float* __restrict__ Wr1,
                             const float* __restrict__ Wl2, const float* __restrict__ Wr2,
                             unsigned short* __restrict__ WT1, unsigned short* __restrict__ WT2) {
    int t = blockIdx.x * blockDim.x + threadIdx.x;
    if (t < 128 * 128) {
        int col = t / 128, k = t % 128;
        float v = (col < 64) ? Wl1[k * 64 + col] : Wr1[k * 64 + (col - 64)];
        WT1[t] = f2bf(v);
    } else if (t < 128 * 128 + 64 * 64) {
        int t2 = t - 128 * 128;
        int col = t2 / 64, k = t2 % 64;
        float v = (col < 32) ? Wl2[k * 32 + col] : Wr2[k * 32 + (col - 32)];
        WT2[t2] = f2bf(v);
    }
}

// ---------------- WMMA GEMM:  out[n x NW] = A[n x K](bf16) @ WT(col-major bf16) ----------------
// blockDim = 256 (8 waves); NT = NW/16 tiles along N, STRIPS row-strips of 16 per block.
template <int K, int NW, int STRIPS>
__global__ void gemm_wmma_kernel(const unsigned short* __restrict__ Abf,
                                 const unsigned short* __restrict__ WT,
                                 float* __restrict__ out, int nrows) {
    constexpr int NT = NW / 16;
    static_assert(STRIPS * NT * 32 == 256, "blockDim mismatch");
    constexpr int NU = (STRIPS * 16 * K) / 2;            // uints in the A tile (1024 -> 4KB)
    __shared__ __align__(16) unsigned int ldsA[NU];

    const int tid   = threadIdx.x;
    const int wave  = tid >> 5;
    const int lane  = tid & 31;
    const int strip = wave / NT;
    const int ntile = wave % NT;
    const int blockRow0 = blockIdx.x * (STRIPS * 16);

    const unsigned int* gsrc =
        reinterpret_cast<const unsigned int*>(Abf) + (size_t)blockRow0 * (K / 2);

#if __has_builtin(__builtin_amdgcn_tensor_load_to_lds) && __has_builtin(__builtin_amdgcn_s_wait_tensorcnt)
    // Stage the 4KB A tile with the Tensor Data Mover (1-D D# descriptor).
    if (wave == 0) {
        unsigned long long gaddr =
            (unsigned long long)(const void*)gsrc;
        unsigned lds_byte =
            (unsigned)(unsigned long long)(__attribute__((address_space(3))) char*)ldsA;
        u32x4 g0 = { 1u,                                   // count=1, user descriptor
                     lds_byte,                             // lds_addr
                     (unsigned)(gaddr & 0xFFFFFFFFull),    // global_addr[31:0]
                     (unsigned)((gaddr >> 32) & 0x1FFFFFFull) | (2u << 30) };  // addr[56:32] | type=2
        i32x8 g1 = { (int)(2u << 16),          // data_size=2 (4B elements)
                     (int)(NU << 16),          // tensor_dim0 = NU (bits 79:48)
                     (int)(1u << 16),          // tensor_dim1 = 1 (bits 111:80)
                     (int)((unsigned)NU << 16),// tile_dim0 = NU (bits 127:112)
                     1,                        // tile_dim1 = 1
                     NU,                       // tensor_dim0_stride (bits 207:160)
                     0, 0 };
        i32x4 gz4 = { 0, 0, 0, 0 };
        i32x8 gz8 = { 0, 0, 0, 0, 0, 0, 0, 0 };
        __builtin_amdgcn_tensor_load_to_lds(g0, g1, gz4, gz4, gz8, 0);
        __builtin_amdgcn_s_wait_tensorcnt(0);
    }
#else
    #pragma unroll
    for (int i = tid; i < NU; i += 256) ldsA[i] = gsrc[i];
#endif
    __syncthreads();

    const int half = (lane < 16) ? 0 : 1;   // lane half selects K sub-range
    const int l16  = lane & 15;
    const int arow = strip * 16 + l16;      // A row within LDS strip (M = l16)
    const int col  = ntile * 16 + l16;      // B column (N = l16)
    const unsigned int* wt = reinterpret_cast<const unsigned int*>(WT);

    v8f c = {};
#pragma unroll
    for (int k0 = 0; k0 < K; k0 += 32) {
        // A frag: lanes<16 hold K {k0..k0+7, k0+16..k0+23}; lanes>=16 hold {+8..+15, +24..+31}
        Frag16 a, b;
        int aidx = (arow * K + k0 + half * 8) >> 1;            // uint index (2 bf16/uint)
        a.q[0] = *reinterpret_cast<const uint4*>(&ldsA[aidx]);
        a.q[1] = *reinterpret_cast<const uint4*>(&ldsA[aidx + 8]);
        // B frag: lane = column, K {k0..k0+7, k0+16..k0+23} (+8 for upper half), same as A
        int bidx = (col * K + k0 + half * 8) >> 1;
        b.q[0] = *reinterpret_cast<const uint4*>(&wt[bidx]);
        b.q[1] = *reinterpret_cast<const uint4*>(&wt[bidx + 8]);

        c = __builtin_amdgcn_wmma_f32_16x16x32_bf16(
                /*neg_a=*/false, a.v, /*neg_b=*/false, b.v,
                /*c_mod=*/(short)0, c, /*reuse_a=*/false, /*reuse_b=*/false);
    }

    // C/D layout: VGPR v -> M = v (lanes 0-15) or v+8 (lanes 16-31), N = l16
    const int baseRow = blockRow0 + strip * 16;
    float* orow = out + (size_t)baseRow * NW + ntile * 16 + l16;
    if (baseRow + 16 <= nrows) {            // full tile: branchless stores
#pragma unroll
        for (int v = 0; v < 8; ++v)
            orow[(size_t)(v + half * 8) * NW] = c[v];
    } else {                                // tail tile (never taken for n%16==0)
#pragma unroll
        for (int v = 0; v < 8; ++v) {
            int row = baseRow + v + half * 8;
            if (row < nrows) out[(size_t)row * NW + ntile * 16 + l16] = c[v];
        }
    }
}

// ---------------- edge scatter:  agg[dst] += P[src] (first D cols of LD-wide rows) --------------
template <int D, int LD>
__global__ void scatter_kernel(const int* __restrict__ ei, const float* __restrict__ P,
                               float* __restrict__ agg, int E) {
    unsigned int t = blockIdx.x * blockDim.x + threadIdx.x;
    if (t < (unsigned int)E * D) {
        int e = t / D;
        int f = t % D;
        int s = ei[e];
        int d = ei[E + e];
        atomicAdd(&agg[(size_t)d * D + f], P[(size_t)s * LD + f]);
    }
}

// ---------------- epilogue: lrelu(agg*deg_inv + R + b); write bf16 (hidden) or f32 (output) ----
template <int D, int LD, bool BF>
__global__ void combine_kernel(const float* __restrict__ agg, const float* __restrict__ out,
                               const float* __restrict__ deginv, const float* __restrict__ bias,
                               int n, float* __restrict__ resF, unsigned short* __restrict__ resB) {
    int t = blockIdx.x * blockDim.x + threadIdx.x;
    if (t < n * D) {
        int nn = t / D, f = t % D;
        float v = agg[t] * deginv[nn] + out[(size_t)nn * LD + D + f] + bias[f];
        v = lrelu(v);
        if (BF) resB[t] = f2bf(v);
        else    resF[t] = v;
    }
}

extern "C" void kernel_launch(void* const* d_in, const int* in_sizes, int n_in,
                              void* d_out, int out_size, void* d_ws, size_t ws_size,
                              hipStream_t stream) {
    const float* x   = (const float*)d_in[0];
    const int*   ei  = (const int*)d_in[1];          // [2][E]
    const float* Wl1 = (const float*)d_in[2];
    const float* Wr1 = (const float*)d_in[3];
    const float* b1  = (const float*)d_in[4];
    const float* Wl2 = (const float*)d_in[5];
    const float* Wr2 = (const float*)d_in[6];
    const float* b2  = (const float*)d_in[7];

    const int n = in_sizes[0] / 128;   // 100000
    const int E = in_sizes[1] / 2;     // 1600000

    // workspace layout (floats), regions reused across layers (~103 MB peak)
    float* ws = (float*)d_ws;
    const size_t Nsz = (size_t)n;
    float*          deg  = ws;                                       // [N]
    unsigned short* x16  = (unsigned short*)(ws + Nsz);              // [N][128] bf16 (reused as h16)
    float*          out1 = ws + Nsz * 65;                            // [N][128] f32 (reused as out2)
    float*          agg  = ws + Nsz * 193;                           // [N][64]  f32 (reused layer 2)
    unsigned short* WT1  = (unsigned short*)(ws + Nsz * 257);        // 128x128 bf16
    unsigned short* WT2  = WT1 + 128 * 128;                          // 64x64 bf16
    unsigned short* h16  = x16;
    float*          out2 = out1;

    const int T = 256;

    // degree + reciprocal
    (void)hipMemsetAsync(deg, 0, Nsz * sizeof(float), stream);
    deg_kernel<<<(E + T - 1) / T, T, 0, stream>>>(ei, deg, E);
    inv_kernel<<<(n + T - 1) / T, T, 0, stream>>>(deg, n);

    // bf16 conversions
    cvt_x_kernel<<<((n * 128) + T - 1) / T, T, 0, stream>>>(x, x16, n * 128);
    cvt_w_kernel<<<(128 * 128 + 64 * 64 + T - 1) / T, T, 0, stream>>>(Wl1, Wr1, Wl2, Wr2, WT1, WT2);

    // layer 1: fused [P1|R1] = x @ [W_l1|W_r1]  -> out1 [N][128]
    gemm_wmma_kernel<128, 128, 1><<<(n + 15) / 16, T, 0, stream>>>(x16, WT1, out1, n);
    (void)hipMemsetAsync(agg, 0, Nsz * 64 * sizeof(float), stream);
    scatter_kernel<64, 128><<<((unsigned)E * 64 + T - 1) / T, T, 0, stream>>>(ei, out1, agg, E);
    combine_kernel<64, 128, true><<<((n * 64) + T - 1) / T, T, 0, stream>>>(
        agg, out1, deg, b1, n, nullptr, h16);

    // layer 2: fused [P2|R2] = h @ [W_l2|W_r2]  -> out2 [N][64]
    gemm_wmma_kernel<64, 64, 2><<<(n + 31) / 32, T, 0, stream>>>(h16, WT2, out2, n);
    (void)hipMemsetAsync(agg, 0, Nsz * 32 * sizeof(float), stream);
    scatter_kernel<32, 64><<<((unsigned)E * 32 + T - 1) / T, T, 0, stream>>>(ei, out2, agg, E);
    combine_kernel<32, 64, false><<<((n * 32) + T - 1) / T, T, 0, stream>>>(
        agg, out2, deg, b2, n, (float*)d_out, nullptr);
}
